// QuantWeightLeNet_90383291777397
// MI455X (gfx1250) — 8-bit matrix operand order repaired and checked by reading, compiled, not run on hardware
//
#include <hip/hip_runtime.h>
#include <stdint.h>

typedef int v8i __attribute__((ext_vector_type(8)));

__global__ void quant_prep(const float* __restrict__ w, int nw, float maxlev,
                           float* scale_out, int8_t* codes, float* wq_f32,
                           const float* __restrict__ b, int nb,
                           const float* __restrict__ sprev, float* bq_out) {
  __shared__ float red[256];
  __shared__ float ssh;
  int tid = threadIdx.x;
  float m = 0.f;
  for (int i = tid; i < nw; i += 256) m = fmaxf(m, fabsf(w[i]));
  red[tid] = m;
  __syncthreads();
  for (int s = 128; s > 0; s >>= 1) {
    if (tid < s) red[tid] = fmaxf(red[tid], red[tid + s]);
    __syncthreads();
  }
  if (tid == 0) {
    float sc = red[0] / maxlev;
    ssh = sc;
    if (scale_out) *scale_out = sc;
  }
  __syncthreads();
  float sc = ssh;
  for (int i = tid; i < nw; i += 256) {
    float q = fminf(fmaxf(rintf(w[i] / sc), -maxlev), maxlev);
    if (codes)  codes[i]  = (int8_t)(int)q;
    if (wq_f32) wq_f32[i] = q * sc;
  }
  if (b && tid < nb) {
    float step = sprev[0] * (1.f / 15.f);
    float sb = sc * step;
    bq_out[tid] = rintf(b[tid] / sb) * sb;
  }
}

__global__ void pack_b_kernel(const int8_t* __restrict__ Wc, int N, int K,
                              int nks, int* __restrict__ out) {
  int tileN = blockIdx.x;
  int ksi = blockIdx.y;
  int lane = threadIdx.x;
  int half = lane >> 4;
  int col = tileN * 16 + (lane & 15);
#pragma unroll
  for (int v = 0; v < 8; ++v) {
    int kb = ksi * 64 + ((v >> 2) << 5) + ((v & 3) << 2) + (half << 4);
    int d = 0;
#pragma unroll
    for (int j = 0; j < 4; ++j) {
      int k = kb + j;
      int w = (col < N && k < K) ? (int)Wc[(long long)col * K + k] : 0;
      d |= (w & 0xFF) << (8 * j);
    }
    out[(((long long)tileN * nks + ksi) * 32 + lane) * 8 + v] = d;
  }
}

__global__ void conv1_kernel(const float* __restrict__ x,
                             const float* __restrict__ wq,
                             const float* __restrict__ s1,
                             uint8_t* __restrict__ out, int B) {
  __shared__ float wsh[162];
  if (threadIdx.x < 162) wsh[threadIdx.x] = wq[threadIdx.x];
  __syncthreads();

  long long t = (long long)blockIdx.x * 256 + threadIdx.x;
  const long long total = (long long)B * 6 * 79 * 79;
  if (t >= total) return;
  int ox = (int)(t % 79); t /= 79;
  int oy = (int)(t % 79); t /= 79;
  int oc = (int)(t % 6);
  int b  = (int)(t / 6);

  float inv_step = 15.f / s1[0];
  int code = 0;
#pragma unroll
  for (int py = 0; py < 2; ++py) {
#pragma unroll
    for (int px = 0; px < 2; ++px) {
      float v = 0.f;
#pragma unroll
      for (int c = 0; c < 3; ++c)
#pragma unroll
        for (int ky = 0; ky < 3; ++ky)
#pragma unroll
          for (int kx = 0; kx < 3; ++kx) {
            int iy = 2 * oy + py + ky;
            int ix = 2 * ox + px + kx;
            v += x[(((long long)b * 3 + c) * 160 + iy) * 160 + ix] *
                 wsh[((oc * 3 + c) * 3 + ky) * 3 + kx];
          }
      int q = (int)fminf(fmaxf(rintf(fmaxf(v, 0.f) * inv_step), 0.f), 15.f);
      code = max(code, q);
    }
  }
  out[(((long long)b * 6 + oc) * 79 + oy) * 79 + ox] = (uint8_t)code;
}

template <int CIN, int COUT>
__global__ void conv_int(const uint8_t* __restrict__ in,
                         const int8_t* __restrict__ wc,
                         const float* __restrict__ scale_w,
                         const float* __restrict__ s_in,
                         const float* __restrict__ bq,
                         const float* __restrict__ s_out,
                         uint8_t* __restrict__ out, int B, int H, int OH,
                         int outBatchStride) {
  __shared__ int8_t wsh[COUT * CIN * 9];
  for (int i = threadIdx.x; i < COUT * CIN * 9; i += 256) wsh[i] = wc[i];
  __syncthreads();

  long long t = (long long)blockIdx.x * 256 + threadIdx.x;
  const long long total = (long long)B * COUT * OH * OH;
  if (t >= total) return;
  int ox = (int)(t % OH); t /= OH;
  int oy = (int)(t % OH); t /= OH;
  int oc = (int)(t % COUT);
  int b  = (int)(t / COUT);

  float sc = scale_w[0] * (s_in[0] * (1.f / 15.f));
  float inv_step = 15.f / s_out[0];
  float bias = bq[oc];
  int code = 0;
#pragma unroll
  for (int py = 0; py < 2; ++py) {
#pragma unroll
    for (int px = 0; px < 2; ++px) {
      int acc = 0;
#pragma unroll
      for (int c = 0; c < CIN; ++c)
#pragma unroll
        for (int ky = 0; ky < 3; ++ky)
#pragma unroll
          for (int kx = 0; kx < 3; ++kx) {
            int iy = 2 * oy + py + ky;
            int ix = 2 * ox + px + kx;
            acc += (int)in[(((long long)b * CIN + c) * H + iy) * H + ix] *
                   (int)wsh[((oc * CIN + c) * 3 + ky) * 3 + kx];
          }
      float y = (float)acc * sc + bias;
      int q = (int)fminf(fmaxf(rintf(fmaxf(y, 0.f) * inv_step), 0.f), 15.f);
      code = max(code, q);
    }
  }
  out[(long long)b * outBatchStride + (oc * OH + oy) * OH + ox] = (uint8_t)code;
}

template <int KPAD>
__global__ void fc_wmma_iu8(const uint8_t* __restrict__ A,
                            const int* __restrict__ Bpack, int N,
                            const float* __restrict__ sw_ptr,
                            const float* __restrict__ sprev_ptr,
                            const float* __restrict__ bq,
                            const float* __restrict__ sout_ptr,
                            uint8_t* __restrict__ out_codes, int outStride,
                            float* __restrict__ out_f32, int Nreal) {
  constexpr int NKS = KPAD / 64;
  int tileM = blockIdx.x, tileN = blockIdx.y;
  int lane = threadIdx.x;
  int half = lane >> 4;
  int l16  = lane & 15;
  int row = tileM * 16 + l16;
  int col = tileN * 16 + l16;

  const uint8_t* arow = A + (long long)row * KPAD;
  const v8i* bfrag = reinterpret_cast<const v8i*>(
      Bpack + ((long long)tileN * NKS * 32 + lane) * 8);

  v8i acc = {};
#pragma unroll
  for (int ksi = 0; ksi < NKS; ++ksi) {
    int ks = ksi * 64;
    v8i a;
#pragma unroll
    for (int v = 0; v < 8; ++v) {
      int ka = ks + ((v >> 2) << 5) + ((v & 3) << 2) + (half << 4);
      a[v] = *reinterpret_cast<const int*>(arow + ka);
    }
    v8i bf = bfrag[ksi * 32];
    acc = __builtin_amdgcn_wmma_i32_16x16x64_iu8(false, a, true, bf, acc,
                                                 false, false);
  }

  float scale = sw_ptr[0] * (sprev_ptr[0] * (1.f / 15.f));
  float colBias = (bq && col < N) ? bq[col] : 0.f;
  if (out_codes) {
    float inv_step = 15.f / sout_ptr[0];
#pragma unroll
    for (int r = 0; r < 8; ++r) {
      int rowD = tileM * 16 + half * 8 + r;
      if (col < N) {
        float y = (float)acc[r] * scale + colBias;
        int q = (int)fminf(fmaxf(rintf(fmaxf(y, 0.f) * inv_step), 0.f), 15.f);
        out_codes[(long long)rowD * outStride + col] = (uint8_t)q;
      }
    }
  } else {
#pragma unroll
    for (int r = 0; r < 8; ++r) {
      int rowD = tileM * 16 + half * 8 + r;
      if (col < Nreal) {
        float y = (float)acc[r] * scale + colBias;
        out_f32[(long long)rowD * Nreal + col] = y;
      }
    }
  }
}

extern "C" void kernel_launch(void* const* d_in, const int* in_sizes, int n_in,
                              void* d_out, int out_size, void* d_ws, size_t ws_size,
                              hipStream_t stream) {
  (void)in_sizes; (void)n_in; (void)out_size; (void)ws_size;
  const float* x   = (const float*)d_in[0];
  const float* w1  = (const float*)d_in[1];
  const float* w2  = (const float*)d_in[2];
  const float* b2  = (const float*)d_in[3];
  const float* w3  = (const float*)d_in[4];
  const float* b3  = (const float*)d_in[5];
  const float* w4  = (const float*)d_in[6];
  const float* b4  = (const float*)d_in[7];
  const float* w5  = (const float*)d_in[8];
  const float* b5  = (const float*)d_in[9];
  const float* fw1 = (const float*)d_in[10];
  const float* fb1 = (const float*)d_in[11];
  const float* fw2 = (const float*)d_in[12];
  const float* fb2 = (const float*)d_in[13];
  const float* fw3 = (const float*)d_in[14];
  const float* s1  = (const float*)d_in[15];
  const float* s2  = (const float*)d_in[16];
  const float* s3  = (const float*)d_in[17];
  const float* s4  = (const float*)d_in[18];
  const float* s5  = (const float*)d_in[19];
  const float* s6  = (const float*)d_in[20];
  const float* s7  = (const float*)d_in[21];

  const int B = 256;

  uintptr_t p = (uintptr_t)d_ws;
  auto alloc = [&](size_t bytes) -> void* {
    void* r = (void*)p;
    p += (bytes + 255) & ~(size_t)255;
    return r;
  };
  float*   scales = (float*)alloc(8 * sizeof(float));
  float*   q1f    = (float*)alloc(162 * sizeof(float));
  int8_t*  w2c    = (int8_t*)alloc(12 * 6 * 9);
  int8_t*  w3c    = (int8_t*)alloc(18 * 12 * 9);
  int8_t*  w4c    = (int8_t*)alloc(18 * 18 * 9);
  int8_t*  w5c    = (int8_t*)alloc(18 * 18 * 9);
  int8_t*  f1c    = (int8_t*)alloc(120 * 162);
  int8_t*  f2c    = (int8_t*)alloc(84 * 120);
  int8_t*  f3c    = (int8_t*)alloc(9 * 84);
  float*   bq2    = (float*)alloc(12 * sizeof(float));
  float*   bq3    = (float*)alloc(18 * sizeof(float));
  float*   bq4    = (float*)alloc(18 * sizeof(float));
  float*   bq5    = (float*)alloc(18 * sizeof(float));
  float*   bqf1   = (float*)alloc(120 * sizeof(float));
  float*   bqf2   = (float*)alloc(84 * sizeof(float));
  int*     bp1    = (int*)alloc((size_t)8 * 3 * 32 * 8 * 4);
  int*     bp2    = (int*)alloc((size_t)6 * 2 * 32 * 8 * 4);
  int*     bp3    = (int*)alloc((size_t)1 * 2 * 32 * 8 * 4);
  uint8_t* act1   = (uint8_t*)alloc((size_t)B * 6 * 79 * 79);
  uint8_t* act2   = (uint8_t*)alloc((size_t)B * 12 * 38 * 38);
  uint8_t* act3   = (uint8_t*)alloc((size_t)B * 18 * 18 * 18);
  uint8_t* act4   = (uint8_t*)alloc((size_t)B * 18 * 8 * 8);
  uint8_t* act5p  = (uint8_t*)alloc((size_t)B * 192);
  uint8_t* fc1o   = (uint8_t*)alloc((size_t)B * 128);
  uint8_t* fc2o   = (uint8_t*)alloc((size_t)B * 128);

  hipMemsetAsync(act5p, 0, (size_t)B * 192, stream);
  hipMemsetAsync(fc1o,  0, (size_t)B * 128, stream);
  hipMemsetAsync(fc2o,  0, (size_t)B * 128, stream);

  quant_prep<<<1, 256, 0, stream>>>(w1, 162, 3.f, scales + 0, nullptr, q1f,
                                    nullptr, 0, nullptr, nullptr);
  quant_prep<<<1, 256, 0, stream>>>(w2, 12 * 6 * 9, 3.f, scales + 1, w2c, nullptr,
                                    b2, 12, s1, bq2);
  quant_prep<<<1, 256, 0, stream>>>(w3, 18 * 12 * 9, 3.f, scales + 2, w3c, nullptr,
                                    b3, 18, s2, bq3);
  quant_prep<<<1, 256, 0, stream>>>(w4, 18 * 18 * 9, 3.f, scales + 3, w4c, nullptr,
                                    b4, 18, s3, bq4);
  quant_prep<<<1, 256, 0, stream>>>(w5, 18 * 18 * 9, 3.f, scales + 4, w5c, nullptr,
                                    b5, 18, s4, bq5);
  quant_prep<<<1, 256, 0, stream>>>(fw1, 120 * 162, 3.f, scales + 5, f1c, nullptr,
                                    fb1, 120, s5, bqf1);
  quant_prep<<<1, 256, 0, stream>>>(fw2, 84 * 120, 3.f, scales + 6, f2c, nullptr,
                                    fb2, 84, s6, bqf2);
  quant_prep<<<1, 256, 0, stream>>>(fw3, 9 * 84, 3.f, scales + 7, f3c, nullptr,
                                    nullptr, 0, nullptr, nullptr);

  pack_b_kernel<<<dim3(8, 3), 32, 0, stream>>>(f1c, 120, 162, 3, bp1);
  pack_b_kernel<<<dim3(6, 2), 32, 0, stream>>>(f2c, 84, 120, 2, bp2);
  pack_b_kernel<<<dim3(1, 2), 32, 0, stream>>>(f3c, 9, 84, 2, bp3);

  {
    long long total = (long long)B * 6 * 79 * 79;
    int blocks = (int)((total + 255) / 256);
    conv1_kernel<<<blocks, 256, 0, stream>>>(x, q1f, s1, act1, B);
  }
  {
    long long total = (long long)B * 12 * 38 * 38;
    conv_int<6, 12><<<(int)((total + 255) / 256), 256, 0, stream>>>(
        act1, w2c, scales + 1, s1, bq2, s2, act2, B, 79, 38, 12 * 38 * 38);
  }
  {
    long long total = (long long)B * 18 * 18 * 18;
    conv_int<12, 18><<<(int)((total + 255) / 256), 256, 0, stream>>>(
        act2, w3c, scales + 2, s2, bq3, s3, act3, B, 38, 18, 18 * 18 * 18);
  }
  {
    long long total = (long long)B * 18 * 8 * 8;
    conv_int<18, 18><<<(int)((total + 255) / 256), 256, 0, stream>>>(
        act3, w4c, scales + 3, s3, bq4, s4, act4, B, 18, 8, 18 * 8 * 8);
  }
  {
    long long total = (long long)B * 18 * 3 * 3;
    conv_int<18, 18><<<(int)((total + 255) / 256), 256, 0, stream>>>(
        act4, w5c, scales + 4, s4, bq5, s5, act5p, B, 8, 3, 192  );
  }

  fc_wmma_iu8<192><<<dim3(16, 8), 32, 0, stream>>>(act5p, bp1, 120,
                                                   scales + 5, s5, bqf1, s6,
                                                   fc1o, 128, nullptr, 0);
  fc_wmma_iu8<128><<<dim3(16, 6), 32, 0, stream>>>(fc1o, bp2, 84,
                                                   scales + 6, s6, bqf2, s7,
                                                   fc2o, 128, nullptr, 0);
  fc_wmma_iu8<128><<<dim3(16, 1), 32, 0, stream>>>(fc2o, bp3, 9,
                                                   scales + 7, s7, nullptr,
                                                   nullptr, nullptr, 0,
                                                   (float*)d_out, 9);
}
